// ShapeEncoder_50732153700824
// MI455X (gfx1250) — matrix operand order repaired and checked by reading, not compiled, not run on hardware
//
#include <hip/hip_runtime.h>
#include <math.h>

#define NNODE 10000
#define NEDGE 160000
#define ETOT  (NEDGE + NNODE)
#define BGR   64
#define HIDC  128
#define HDC   512
#define NHEAD 4
#define EMB   256
#define KPAD  544           // 514 padded up to a multiple of 32

#define BM 128
#define BN 64
#define BK 32
#define LDSTR 40            // padded LDS row stride (ushorts): 16B-aligned, conflict-reducing

typedef __attribute__((ext_vector_type(16))) __bf16 v16bf;
typedef __attribute__((ext_vector_type(8)))  float  v8f;

// ---------- helpers ----------
__device__ inline unsigned short f2bf(float f) {
  union { float f; unsigned u; } x; x.f = f;
  unsigned r = x.u + 0x7fffu + ((x.u >> 16) & 1u);   // round-to-nearest-even
  return (unsigned short)(r >> 16);
}

__device__ inline unsigned enc_ord(float v) {        // order-preserving float->u32
  unsigned b = __float_as_uint(v);
  return (b & 0x80000000u) ? ~b : (b | 0x80000000u);
}
__device__ inline float dec_ord(unsigned e) {
  return (e & 0x80000000u) ? __uint_as_float(e & 0x7fffffffu) : __uint_as_float(~e);
}

__device__ inline void edge_ends(const int* __restrict__ ei, int e, int& s, int& d) {
  if (e < NEDGE) { s = ei[e]; d = ei[NEDGE + e]; }
  else           { s = e - NEDGE; d = e - NEDGE; }   // appended self loops
}

// ---------- WMMA bf16 GEMM: C[M,N] = act(A[M,K] @ W[K,N] + bias) ----------
// Requires K % 32 == 0 (and K >= 32), N % 64 == 0. block = 256 threads (8 waves);
// tile 128x64; each wave computes 32x32 (4 WMMAs / K-step). Global loads for the
// next K-tile are issued through registers while the current tile's WMMAs run.
__global__ __launch_bounds__(256) void wmma_gemm_kernel(
    const float* __restrict__ A, const float* __restrict__ W,
    const float* __restrict__ bias, float* __restrict__ C,
    int M, int K, int N, int act)
{
  __shared__ __align__(16) unsigned short sA[BM * LDSTR];  // [row][k]
  __shared__ __align__(16) unsigned short sB[BN * LDSTR];  // [n][k] (transposed W tile)
  const int tid  = threadIdx.x;
  const int wave = tid >> 5, lane = tid & 31;
  const int m0 = blockIdx.x * BM;
  const int n0 = blockIdx.y * BN;
  const int rowblk = wave >> 1;   // 0..3 -> 32-row slab
  const int colblk = wave & 1;    // 0..1 -> 32-col slab
  v8f acc00 = {}, acc01 = {}, acc10 = {}, acc11 = {};

  // ---- loop-invariant staging geometry ----
  // A side: 4 float4 per thread (128x32 tile)
  const float* pA[4];
  unsigned short* dA[4];
#pragma unroll
  for (int t = 0; t < 4; ++t) {
    int f = tid + t * 256;                  // 0..1023
    int r = f >> 3, k = (f & 7) << 2;
    int gm = m0 + r; if (gm > M - 1) gm = M - 1;   // clamp tail rows (masked at store)
    pA[t] = A + (size_t)gm * K + k;
    dA[t] = sA + r * LDSTR + k;
  }
  // B side: 2 float4 per thread (32x64 tile, stored transposed)
  const float* pB[2];
  int kB[2], nB[2];
#pragma unroll
  for (int t = 0; t < 2; ++t) {
    int f = tid + t * 256;                  // 0..511
    int k = f >> 4, n = (f & 15) << 2;
    kB[t] = k; nB[t] = n;
    pB[t] = W + (size_t)k * N + n0 + n;
  }
  // fragment LDS addresses (invariant): A lane = row (lane&15), K chunks at
  // (lane>>4)*8 and +16; B lane = col (lane&15), K chunks at (lane>>4)*8 and +16.
  const int ar  = rowblk * 32 + (lane & 15);
  const int akb = (lane >> 4) * 8;
  const unsigned short* fraA0 = sA + ar * LDSTR + akb;
  const unsigned short* fraA1 = sA + (ar + 16) * LDSTR + akb;
  const int bkb = (lane >> 4) * 8;
  const int bc  = colblk * 32 + (lane & 15);
  const unsigned short* frbB0 = sB + bc * LDSTR + bkb;
  const unsigned short* frbB1 = sB + (bc + 16) * LDSTR + bkb;

  // ---- prologue: fetch tile k0 = 0 into registers ----
  float4 ra[4], rb[2];
#pragma unroll
  for (int t = 0; t < 4; ++t) ra[t] = *(const float4*)pA[t];
#pragma unroll
  for (int t = 0; t < 2; ++t) rb[t] = *(const float4*)pB[t];

  for (int k0 = 0; k0 < K; k0 += BK) {
    // convert current registers -> LDS (bf16)
#pragma unroll
    for (int t = 0; t < 4; ++t) {
      unsigned short* d = dA[t];
      d[0] = f2bf(ra[t].x); d[1] = f2bf(ra[t].y);
      d[2] = f2bf(ra[t].z); d[3] = f2bf(ra[t].w);
    }
#pragma unroll
    for (int t = 0; t < 2; ++t) {
      sB[(nB[t] + 0) * LDSTR + kB[t]] = f2bf(rb[t].x);
      sB[(nB[t] + 1) * LDSTR + kB[t]] = f2bf(rb[t].y);
      sB[(nB[t] + 2) * LDSTR + kB[t]] = f2bf(rb[t].z);
      sB[(nB[t] + 3) * LDSTR + kB[t]] = f2bf(rb[t].w);
    }
    __syncthreads();

    // issue next tile's global loads; results not used until next iteration's
    // convert phase, so the loadcnt wait lands after the WMMAs below.
    if (k0 + BK < K) {
#pragma unroll
      for (int t = 0; t < 4; ++t) { pA[t] += BK; ra[t] = *(const float4*)pA[t]; }
#pragma unroll
      for (int t = 0; t < 2; ++t) { pB[t] += (size_t)BK * N; rb[t] = *(const float4*)pB[t]; }
    }

    union { v16bf v; uint4 q[2]; } fa0, fa1, fb0, fb1;
    fa0.q[0] = *(const uint4*)(fraA0);
    fa0.q[1] = *(const uint4*)(fraA0 + 16);
    fa1.q[0] = *(const uint4*)(fraA1);
    fa1.q[1] = *(const uint4*)(fraA1 + 16);
    fb0.q[0] = *(const uint4*)(frbB0);
    fb0.q[1] = *(const uint4*)(frbB0 + 16);
    fb1.q[0] = *(const uint4*)(frbB1);
    fb1.q[1] = *(const uint4*)(frbB1 + 16);

    acc00 = __builtin_amdgcn_wmma_f32_16x16x32_bf16(false, fa0.v, false, fb0.v,
                                                    (short)0, acc00, false, false);
    acc01 = __builtin_amdgcn_wmma_f32_16x16x32_bf16(false, fa0.v, false, fb1.v,
                                                    (short)0, acc01, false, false);
    acc10 = __builtin_amdgcn_wmma_f32_16x16x32_bf16(false, fa1.v, false, fb0.v,
                                                    (short)0, acc10, false, false);
    acc11 = __builtin_amdgcn_wmma_f32_16x16x32_bf16(false, fa1.v, false, fb1.v,
                                                    (short)0, acc11, false, false);
    __syncthreads();
  }

  // C/D layout: VGPR j -> M = j + 8*(lane>=16), N = lane&15
  int rbse = m0 + rowblk * 32 + ((lane >> 4) << 3);
  int cb   = n0 + colblk * 32 + (lane & 15);
  float b0 = bias ? bias[cb]      : 0.0f;
  float b1 = bias ? bias[cb + 16] : 0.0f;
#pragma unroll
  for (int j = 0; j < 8; ++j) {
    int row0 = rbse + j, row1 = rbse + 16 + j;
    if (row0 < M) {
      float v0 = acc00[j] + b0, v1 = acc01[j] + b1;
      if (act == 1) { v0 = fmaxf(v0, 0.0f); v1 = fmaxf(v1, 0.0f); }
      C[(size_t)row0 * N + cb]      = v0;
      C[(size_t)row0 * N + cb + 16] = v1;
    }
    if (row1 < M) {
      float v0 = acc10[j] + b0, v1 = acc11[j] + b1;
      if (act == 1) { v0 = fmaxf(v0, 0.0f); v1 = fmaxf(v1, 0.0f); }
      C[(size_t)row1 * N + cb]      = v0;
      C[(size_t)row1 * N + cb + 16] = v1;
    }
  }
}

// ---------- small kernels ----------
__global__ void fill_f32(float* p, float v, int n) {
  int i = blockIdx.x * blockDim.x + threadIdx.x;
  if (i < n) p[i] = v;
}

__global__ void enc1_kernel(const float* __restrict__ x, const float* __restrict__ w1,
                            const float* __restrict__ b1, float* __restrict__ out) {
  int i = blockIdx.x * blockDim.x + threadIdx.x;
  if (i >= NNODE * HIDC) return;
  int node = i / HIDC, j = i % HIDC;
  float v = x[node * 3 + 0] * w1[0 * HIDC + j]
          + x[node * 3 + 1] * w1[1 * HIDC + j]
          + x[node * 3 + 2] * w1[2 * HIDC + j] + b1[j];
  out[i] = fmaxf(v, 0.0f);
}

// one wave per edge: logits[e,h] = att_h . leaky_relu(xl[src]+xr[dst])_h
__global__ __launch_bounds__(256) void edge_logits_kernel(
    const float* __restrict__ xl, const float* __restrict__ xr,
    const int* __restrict__ ei, const float* __restrict__ att,
    float* __restrict__ logits)
{
  int e = blockIdx.x * 8 + (threadIdx.x >> 5);
  if (e >= ETOT) return;
  int lane = threadIdx.x & 31;
  int s, d; edge_ends(ei, e, s, d);
  int c0 = lane * 16;                       // lane covers 16 channels inside one head
  const float4* pl = (const float4*)(xl + (size_t)s * HDC + c0);
  const float4* pr = (const float4*)(xr + (size_t)d * HDC + c0);
  const float4* pa = (const float4*)(att + c0);
  float sum = 0.f;
#pragma unroll
  for (int i = 0; i < 4; ++i) {
    float4 a = pl[i], b = pr[i], w = pa[i];
    float v;
    v = a.x + b.x; v = v > 0.f ? v : 0.2f * v; sum += v * w.x;
    v = a.y + b.y; v = v > 0.f ? v : 0.2f * v; sum += v * w.y;
    v = a.z + b.z; v = v > 0.f ? v : 0.2f * v; sum += v * w.z;
    v = a.w + b.w; v = v > 0.f ? v : 0.2f * v; sum += v * w.w;
  }
  sum += __shfl_xor(sum, 1, 32);
  sum += __shfl_xor(sum, 2, 32);
  sum += __shfl_xor(sum, 4, 32);            // reduce within 8-lane head group
  if ((lane & 7) == 0) logits[(size_t)e * NHEAD + (lane >> 3)] = sum;
}

__global__ void edge_amax_kernel(const float* __restrict__ logits,
                                 const int* __restrict__ ei, unsigned* __restrict__ maxb) {
  int t = blockIdx.x * blockDim.x + threadIdx.x;
  if (t >= ETOT * NHEAD) return;
  int e = t >> 2, h = t & 3;
  int s, d; edge_ends(ei, e, s, d); (void)s;
  atomicMax(&maxb[d * NHEAD + h], enc_ord(logits[t]));
}

__global__ void edge_exp_kernel(const float* __restrict__ logits, const int* __restrict__ ei,
                                const unsigned* __restrict__ maxb,
                                float* __restrict__ eexp, float* __restrict__ denom) {
  int t = blockIdx.x * blockDim.x + threadIdx.x;
  if (t >= ETOT * NHEAD) return;
  int e = t >> 2, h = t & 3;
  int s, d; edge_ends(ei, e, s, d); (void)s;
  float m  = dec_ord(maxb[d * NHEAD + h]);
  float ee = __expf(logits[t] - m);
  eexp[t] = ee;
  atomicAdd(&denom[d * NHEAD + h], ee);
}

// one wave per edge: agg[dst] += alpha * xl[src]
__global__ __launch_bounds__(256) void edge_agg_kernel(
    const float* __restrict__ xl, const float* __restrict__ eexp,
    const float* __restrict__ denom, const int* __restrict__ ei,
    float* __restrict__ agg)
{
  int e = blockIdx.x * 8 + (threadIdx.x >> 5);
  if (e >= ETOT) return;
  int lane = threadIdx.x & 31;
  int s, d; edge_ends(ei, e, s, d);
  int h = lane >> 3;
  float alpha = eexp[(size_t)e * NHEAD + h] / denom[d * NHEAD + h];
  int c0 = lane * 16;
  const float4* pl = (const float4*)(xl + (size_t)s * HDC + c0);
  float* pa = agg + (size_t)d * HDC + c0;
#pragma unroll
  for (int i = 0; i < 4; ++i) {
    float4 v = pl[i];
    atomicAdd(pa + i * 4 + 0, alpha * v.x);
    atomicAdd(pa + i * 4 + 1, alpha * v.y);
    atomicAdd(pa + i * 4 + 2, alpha * v.z);
    atomicAdd(pa + i * 4 + 3, alpha * v.w);
  }
}

// block per node: LayerNorm(agg + bias) * g + b, + res, ELU -> hout
__global__ __launch_bounds__(256) void ln_elu_kernel(
    const float* __restrict__ agg, const float* __restrict__ gb,
    const float* __restrict__ lng, const float* __restrict__ lnb,
    const float* __restrict__ res, float* __restrict__ hout)
{
  __shared__ float s1[256], s2[256];
  int n = blockIdx.x, t = threadIdx.x;
  float a0 = agg[(size_t)n * HDC + t]       + gb[t];
  float a1 = agg[(size_t)n * HDC + t + 256] + gb[t + 256];
  s1[t] = a0 + a1;
  s2[t] = a0 * a0 + a1 * a1;
  __syncthreads();
  for (int o = 128; o > 0; o >>= 1) {
    if (t < o) { s1[t] += s1[t + o]; s2[t] += s2[t + o]; }
    __syncthreads();
  }
  float mean = s1[0] * (1.0f / HDC);
  float var  = s2[0] * (1.0f / HDC) - mean * mean;
  float inv  = rsqrtf(var + 1e-5f);
  {
    int c = t;
    float hn = (a0 - mean) * inv * lng[c] + lnb[c] + res[(size_t)n * HDC + c];
    hout[(size_t)n * HDC + c] = hn > 0.f ? hn : expm1f(hn);
  }
  {
    int c = t + 256;
    float hn = (a1 - mean) * inv * lng[c] + lnb[c] + res[(size_t)n * HDC + c];
    hout[(size_t)n * HDC + c] = hn > 0.f ? hn : expm1f(hn);
  }
}

// wave per node: gate[n] = g1[n,:] . w2 + b2
__global__ __launch_bounds__(256) void gate2_kernel(
    const float* __restrict__ g1, const float* __restrict__ w2,
    const float* __restrict__ b2, float* __restrict__ gate)
{
  int n = blockIdx.x * 8 + (threadIdx.x >> 5);
  if (n >= NNODE) return;
  int lane = threadIdx.x & 31;
  float4 v = *(const float4*)(g1 + (size_t)n * HIDC + lane * 4);
  float4 w = *(const float4*)(w2 + lane * 4);
  float sum = v.x * w.x + v.y * w.y + v.z * w.z + v.w * w.w;
  sum += __shfl_xor(sum, 16, 32); sum += __shfl_xor(sum, 8, 32);
  sum += __shfl_xor(sum, 4, 32);  sum += __shfl_xor(sum, 2, 32);
  sum += __shfl_xor(sum, 1, 32);
  if (lane == 0) gate[n] = sum + b2[0];
}

__global__ void pool_amax_kernel(const float* __restrict__ gate, const int* __restrict__ batch,
                                 unsigned* __restrict__ pmax) {
  int n = blockIdx.x * blockDim.x + threadIdx.x;
  if (n >= NNODE) return;
  atomicMax(&pmax[batch[n]], enc_ord(gate[n]));
}

__global__ void pool_exp_kernel(const float* __restrict__ gate, const int* __restrict__ batch,
                                const float* __restrict__ x, const unsigned* __restrict__ pmax,
                                float* __restrict__ wn, float* __restrict__ pden,
                                float* __restrict__ nnc, float* __restrict__ tb) {
  int n = blockIdx.x * blockDim.x + threadIdx.x;
  if (n >= NNODE) return;
  int b = batch[n];
  float w = __expf(gate[n] - dec_ord(pmax[b]));
  wn[n] = w;
  atomicAdd(&pden[b], w);
  atomicAdd(&nnc[b], 1.0f);
  atomicAdd(&tb[b], x[n * 3 + 1]);
}

__global__ __launch_bounds__(256) void pool_agg_kernel(
    const float* __restrict__ h, const float* __restrict__ wn,
    const float* __restrict__ pden, const int* __restrict__ batch,
    float* __restrict__ g)
{
  int n = blockIdx.x * 8 + (threadIdx.x >> 5);
  if (n >= NNODE) return;
  int lane = threadIdx.x & 31;
  int b = batch[n];
  float w = wn[n] / pden[b];
  int c0 = lane * 16;
  const float4* ph = (const float4*)(h + (size_t)n * HDC + c0);
  float* pg = g + (size_t)b * HDC + c0;
#pragma unroll
  for (int i = 0; i < 4; ++i) {
    float4 v = ph[i];
    atomicAdd(pg + i * 4 + 0, w * v.x);
    atomicAdd(pg + i * 4 + 1, w * v.y);
    atomicAdd(pg + i * 4 + 2, w * v.z);
    atomicAdd(pg + i * 4 + 3, w * v.w);
  }
}

__global__ void gcat_kernel(const float* __restrict__ g, const float* __restrict__ nnc,
                            const float* __restrict__ tb, float* __restrict__ gcat) {
  int i = blockIdx.x * blockDim.x + threadIdx.x;
  if (i >= BGR * KPAD) return;
  int b = i / KPAD, c = i % KPAD;
  float v;
  if (c < HDC)       v = g[(size_t)b * HDC + c];
  else if (c == 512) v = nnc[b] * (1.0f / 20.0f);
  else if (c == 513) v = tb[b] * 0.31830988618379067f;   // 1/pi
  else               v = 0.0f;                           // zero pad to KPAD
  gcat[i] = v;
}

__global__ __launch_bounds__(256) void norm_kernel(const float* __restrict__ z,
                                                   float* __restrict__ out) {
  int r = blockIdx.x * 8 + (threadIdx.x >> 5);
  if (r >= BGR) return;
  int lane = threadIdx.x & 31;
  float ss = 0.f;
#pragma unroll
  for (int i = 0; i < 8; ++i) { float v = z[(size_t)r * EMB + lane * 8 + i]; ss += v * v; }
  ss += __shfl_xor(ss, 16, 32); ss += __shfl_xor(ss, 8, 32);
  ss += __shfl_xor(ss, 4, 32);  ss += __shfl_xor(ss, 2, 32);
  ss += __shfl_xor(ss, 1, 32);
  float inv = rsqrtf(ss);
#pragma unroll
  for (int i = 0; i < 8; ++i) {
    int c = lane * 8 + i;
    out[(size_t)r * EMB + c] = z[(size_t)r * EMB + c] * inv;
  }
}

// ---------- launch ----------
extern "C" void kernel_launch(void* const* d_in, const int* in_sizes, int n_in,
                              void* d_out, int out_size, void* d_ws, size_t ws_size,
                              hipStream_t stream) {
  const float* x      = (const float*)d_in[0];
  const int*   ei     = (const int*)d_in[1];
  const int*   batch  = (const int*)d_in[2];
  const float* enc_w1 = (const float*)d_in[3];
  const float* enc_b1 = (const float*)d_in[4];
  const float* enc_w2 = (const float*)d_in[5];
  const float* enc_b2 = (const float*)d_in[6];
  const float *wl[4], *bl[4], *wr[4], *br[4], *att[4], *gbias[4], *lng[4], *lnb[4];
  for (int l = 0; l < 4; ++l) {
    int base = 7 + l * 8;
    wl[l]    = (const float*)d_in[base + 0];
    bl[l]    = (const float*)d_in[base + 1];
    wr[l]    = (const float*)d_in[base + 2];
    br[l]    = (const float*)d_in[base + 3];
    att[l]   = (const float*)d_in[base + 4];
    gbias[l] = (const float*)d_in[base + 5];
    lng[l]   = (const float*)d_in[base + 6];
    lnb[l]   = (const float*)d_in[base + 7];
  }
  const float* res0_w  = (const float*)d_in[39];
  const float* gate_w1 = (const float*)d_in[40];
  const float* gate_b1 = (const float*)d_in[41];
  const float* gate_w2 = (const float*)d_in[42];
  const float* gate_b2 = (const float*)d_in[43];
  const float* proj_w1 = (const float*)d_in[44];
  const float* proj_b1 = (const float*)d_in[45];
  const float* proj_w2 = (const float*)d_in[46];
  const float* proj_b2 = (const float*)d_in[47];

  char* wp = (char*)d_ws;
  auto carve = [&](size_t nbytes) -> void* {
    void* r = (void*)wp;
    wp += (nbytes + 255) & ~(size_t)255;
    return r;
  };
  float*    hEnc1  = (float*)carve((size_t)NNODE * HIDC * 4);
  float*    hEnc   = (float*)carve((size_t)NNODE * HIDC * 4);
  float*    bufH   = (float*)carve((size_t)NNODE * HDC * 4);
  float*    bufXL  = (float*)carve((size_t)NNODE * HDC * 4);
  float*    bufXR  = (float*)carve((size_t)NNODE * HDC * 4);
  float*    bufRes = (float*)carve((size_t)NNODE * HDC * 4);
  float*    bufAgg = (float*)carve((size_t)NNODE * HDC * 4);
  float*    logits = (float*)carve((size_t)ETOT * NHEAD * 4);
  float*    eexp   = (float*)carve((size_t)ETOT * NHEAD * 4);
  unsigned* maxb   = (unsigned*)carve((size_t)NNODE * NHEAD * 4);
  float*    denom  = (float*)carve((size_t)NNODE * NHEAD * 4);
  float*    gate1  = (float*)carve((size_t)NNODE * HIDC * 4);
  float*    gate   = (float*)carve((size_t)NNODE * 4);
  float*    wn     = (float*)carve((size_t)NNODE * 4);
  unsigned* pmax   = (unsigned*)carve((size_t)BGR * 4);
  float*    pden   = (float*)carve((size_t)BGR * 4);
  float*    nnc    = (float*)carve((size_t)BGR * 4);
  float*    tb     = (float*)carve((size_t)BGR * 4);
  float*    g      = (float*)carve((size_t)BGR * HDC * 4);
  float*    gcat   = (float*)carve((size_t)BGR * KPAD * 4);
  float*    z1     = (float*)carve((size_t)BGR * HDC * 4);
  float*    z2     = (float*)carve((size_t)BGR * EMB * 4);

  // encoder
  enc1_kernel<<<(NNODE * HIDC + 255) / 256, 256, 0, stream>>>(x, enc_w1, enc_b1, hEnc1);
  wmma_gemm_kernel<<<dim3((NNODE + BM - 1) / BM, HIDC / BN), 256, 0, stream>>>(
      hEnc1, enc_w2, enc_b2, hEnc, NNODE, HIDC, HIDC, 0);

  for (int l = 0; l < 4; ++l) {
    const float* in = (l == 0) ? hEnc : bufH;
    int inC = (l == 0) ? HIDC : HDC;
    wmma_gemm_kernel<<<dim3((NNODE + BM - 1) / BM, HDC / BN), 256, 0, stream>>>(
        in, wl[l], bl[l], bufXL, NNODE, inC, HDC, 0);
    wmma_gemm_kernel<<<dim3((NNODE + BM - 1) / BM, HDC / BN), 256, 0, stream>>>(
        in, wr[l], br[l], bufXR, NNODE, inC, HDC, 0);
    const float* resPtr;
    if (l == 0) {
      wmma_gemm_kernel<<<dim3((NNODE + BM - 1) / BM, HDC / BN), 256, 0, stream>>>(
          hEnc, res0_w, nullptr, bufRes, NNODE, HIDC, HDC, 0);
      resPtr = bufRes;
    } else {
      resPtr = bufH;
    }
    fill_f32<<<(NNODE * HDC + 255) / 256, 256, 0, stream>>>(bufAgg, 0.0f, NNODE * HDC);
    fill_f32<<<(NNODE * NHEAD + 255) / 256, 256, 0, stream>>>((float*)maxb, 0.0f, NNODE * NHEAD);
    fill_f32<<<(NNODE * NHEAD + 255) / 256, 256, 0, stream>>>(denom, 0.0f, NNODE * NHEAD);
    edge_logits_kernel<<<(ETOT + 7) / 8, 256, 0, stream>>>(bufXL, bufXR, ei, att[l], logits);
    edge_amax_kernel<<<(ETOT * NHEAD + 255) / 256, 256, 0, stream>>>(logits, ei, maxb);
    edge_exp_kernel<<<(ETOT * NHEAD + 255) / 256, 256, 0, stream>>>(logits, ei, maxb, eexp, denom);
    edge_agg_kernel<<<(ETOT + 7) / 8, 256, 0, stream>>>(bufXL, eexp, denom, ei, bufAgg);
    ln_elu_kernel<<<NNODE, 256, 0, stream>>>(bufAgg, gbias[l], lng[l], lnb[l], resPtr, bufH);
  }

  // attentional pooling
  wmma_gemm_kernel<<<dim3((NNODE + BM - 1) / BM, HIDC / BN), 256, 0, stream>>>(
      bufH, gate_w1, gate_b1, gate1, NNODE, HDC, HIDC, 1);
  gate2_kernel<<<(NNODE + 7) / 8, 256, 0, stream>>>(gate1, gate_w2, gate_b2, gate);
  fill_f32<<<1, 256, 0, stream>>>((float*)pmax, 0.0f, BGR);
  fill_f32<<<1, 256, 0, stream>>>(pden, 0.0f, BGR);
  fill_f32<<<1, 256, 0, stream>>>(nnc, 0.0f, BGR);
  fill_f32<<<1, 256, 0, stream>>>(tb, 0.0f, BGR);
  fill_f32<<<(BGR * HDC + 255) / 256, 256, 0, stream>>>(g, 0.0f, BGR * HDC);
  pool_amax_kernel<<<(NNODE + 255) / 256, 256, 0, stream>>>(gate, batch, pmax);
  pool_exp_kernel<<<(NNODE + 255) / 256, 256, 0, stream>>>(gate, batch, x, pmax, wn, pden, nnc, tb);
  pool_agg_kernel<<<(NNODE + 7) / 8, 256, 0, stream>>>(bufH, wn, pden, batch, g);
  gcat_kernel<<<(BGR * KPAD + 255) / 256, 256, 0, stream>>>(g, nnc, tb, gcat);

  // final projection + normalize (K padded to 544 so K%32==0)
  wmma_gemm_kernel<<<dim3(1, HDC / BN), 256, 0, stream>>>(gcat, proj_w1, proj_b1, z1, BGR, KPAD, HDC, 1);
  wmma_gemm_kernel<<<dim3(1, EMB / BN), 256, 0, stream>>>(z1, proj_w2, proj_b2, z2, BGR, HDC, EMB, 0);
  norm_kernel<<<(BGR + 7) / 8, 256, 0, stream>>>(z2, (float*)d_out);
}
